// MultiheadDiffAttn_11828339933295
// MI455X (gfx1250) — matrix operand order repaired and checked by reading, not compiled, not run on hardware
//
#include <hip/hip_runtime.h>
#include <hip/hip_bf16.h>

// ---------------------------------------------------------------------------
// MultiheadDiffAttn for MI455X (gfx1250, wave32, WMMA bf16 + TDM async copy)
// ---------------------------------------------------------------------------

typedef __attribute__((ext_vector_type(16))) __bf16 v16bf;
typedef __attribute__((ext_vector_type(8)))  float  v8f;
typedef __attribute__((ext_vector_type(4)))  __bf16 v4bf;
typedef __attribute__((ext_vector_type(4)))  unsigned int v4u;
typedef __attribute__((ext_vector_type(4)))  int v4i;
typedef __attribute__((ext_vector_type(8)))  int v8i;

#define TSEQ        2048
#define BATCH       2
#define EMBED       2048
#define MROWS       4096     // BATCH * TSEQ
#define NHEADS      16
#define HD          64
#define DH2         128
#define LAMBDA_INIT 0.9902528748f
#define OUT_SCALE   0.0097471252f   // 1 - LAMBDA_INIT
#define QK_SCALE    0.125f          // HD^-0.5

#define HAVE_TDM __has_builtin(__builtin_amdgcn_tensor_load_to_lds)

// ---------------------------------------------------------------------------
// WMMA helpers (fragment layouts used by this code, wave32)
// A (16x32 bf16): lane r=lane&15 holds row r; element j -> K = j + 16*hi
// B (32x16 bf16): lane n=lane&15 holds col n; element j -> K = j + 16*hi
// C/D (16x16 f32): lane n=lane&15 col; VGPR r -> row r + 8*hi
// ---------------------------------------------------------------------------

union FragU { v16bf v; uint4 q[2]; };
union Pack8 { __bf16 h[8]; uint4 q; };

__device__ inline v16bf lda16(const __bf16* p, int hi) {
    // p = &src[row * ld + k0]; loads halfs k0+16hi..+15 (contiguous)
    FragU u;
    u.q[0] = *(const uint4*)(p + 16 * hi);
    u.q[1] = *(const uint4*)(p + 16 * hi + 8);
    return u.v;
}

__device__ inline v16bf ldb16(const __bf16* p) {
    // p = per-lane pointer at [col n][k = 16*hi]; 16 contiguous halfs
    FragU u;
    u.q[0] = *(const uint4*)p;
    u.q[1] = *(const uint4*)(p + 8);
    return u.v;
}

__device__ inline v8f wmma_bf16(v16bf a, v16bf b, v8f c) {
    return __builtin_amdgcn_wmma_f32_16x16x32_bf16(false, a, false, b,
                                                   (short)0, c, false, false);
}

#if HAVE_TDM
// ---------------------------------------------------------------------------
// Tensor Data Mover: async 2-D tile copy global -> LDS (row-major).
// LDS padding enabled: after every 16 DWORDs (one 32-half tile row) insert
// 4 DWORDs -> LDS row stride of 40 halfs (bank-conflict-avoiding layout).
// Descriptor packing per CDNA5 ISA 8.3/8.4.
// ---------------------------------------------------------------------------
__device__ inline void tdm_load_2d(const __bf16* gptr, unsigned int ldsAddr,
                                   unsigned int tile0, unsigned int tile1,
                                   unsigned int tdim0, unsigned int tdim1,
                                   unsigned int stride0) {
    unsigned long long ga = (unsigned long long)(size_t)gptr;
    v4u g0;
    g0[0] = 1u;                                    // count = 1 (valid)
    g0[1] = ldsAddr;                               // LDS byte address
    g0[2] = (unsigned int)ga;                      // global addr [31:0]
    g0[3] = (unsigned int)(ga >> 32) | (2u << 30); // global addr hi | type=2
    v8i g1;
    g1[0] = (int)((1u << 16)      // data_size = 2 bytes
                | (1u << 20)      // pad_enable
                | (3u << 22)      // pad_interval: 16 DWORDs
                | (3u << 25));    // pad_amount:   4 DWORDs
    g1[1] = (int)((tdim0 & 0xFFFFu) << 16);        // tensor_dim0 [15:0]
    g1[2] = (int)((tdim0 >> 16) | ((tdim1 & 0xFFFFu) << 16));
    g1[3] = (int)((tdim1 >> 16) | (tile0 << 16));  // tile_dim0
    g1[4] = (int)tile1;                            // tile_dim1 (tile_dim2=0)
    g1[5] = (int)stride0;                          // dim0 stride, low 32
    g1[6] = 0;
    g1[7] = 0;
    v4i g2 = {0, 0, 0, 0};
    v4i g3 = {0, 0, 0, 0};
#if __clang_major__ >= 23
    v8i g4 = {0, 0, 0, 0, 0, 0, 0, 0};
    __builtin_amdgcn_tensor_load_to_lds(g0, g1, g2, g3, g4, 0);
#else
    __builtin_amdgcn_tensor_load_to_lds(g0, g1, g2, g3, 0);
#endif
}
#endif  // HAVE_TDM

// ---------------------------------------------------------------------------
// Kernel 1a: f32 -> bf16 conversion (vectorized x4), row-major preserved
// ---------------------------------------------------------------------------
__global__ void cvt_bf16_kernel(const float* __restrict__ s,
                                __bf16* __restrict__ d, int nquads) {
    int i = blockIdx.x * blockDim.x + threadIdx.x;
    if (i >= nquads) return;
    float4 f = ((const float4*)s)[i];
    v4bf o;
    o.x = (__bf16)f.x; o.y = (__bf16)f.y; o.z = (__bf16)f.z; o.w = (__bf16)f.w;
    ((v4bf*)d)[i] = o;
}

// ---------------------------------------------------------------------------
// Kernel 1b: f32 W(KxN) -> bf16 Wt(NxK) transpose via LDS tile.
// Weight transpose makes GEMM B fragments K-contiguous -> TDM-loadable tiles.
// ---------------------------------------------------------------------------
__global__ __launch_bounds__(256)
void cvt_transpose_kernel(const float* __restrict__ W, __bf16* __restrict__ Wt,
                          int K, int N) {
    __shared__ float tile[32][33];
    const int k0 = blockIdx.y * 32;
    const int n0 = blockIdx.x * 32;
    const int tx = threadIdx.x & 31;
    const int ty = threadIdx.x >> 5;   // 0..7
#pragma unroll
    for (int i = 0; i < 4; ++i)
        tile[ty + i * 8][tx] = W[(size_t)(k0 + ty + i * 8) * N + n0 + tx];
    __syncthreads();
#pragma unroll
    for (int i = 0; i < 4; ++i)
        Wt[(size_t)(n0 + ty + i * 8) * K + k0 + tx] = (__bf16)tile[tx][ty + i * 8];
}

// ---------------------------------------------------------------------------
// Kernel 2: lambda scalar
// ---------------------------------------------------------------------------
__global__ void lam_kernel(const float* __restrict__ lq1, const float* __restrict__ lk1,
                           const float* __restrict__ lq2, const float* __restrict__ lk2,
                           float* __restrict__ lamOut) {
    __shared__ float s1[64], s2[64];
    int t = threadIdx.x;
    s1[t] = lq1[t] * lk1[t];
    s2[t] = lq2[t] * lk2[t];
    __syncthreads();
    for (int off = 32; off > 0; off >>= 1) {
        if (t < off) { s1[t] += s1[t + off]; s2[t] += s2[t + off]; }
        __syncthreads();
    }
    if (t == 0) lamOut[0] = __expf(s1[0]) - __expf(s2[0]) + LAMBDA_INIT;
}

// ---------------------------------------------------------------------------
// Kernel 3: bf16 GEMM  C(MxN) = A(MxK) @ Wt^T, with Wt in N x K layout.
// Block 256 threads = 8 waves; tile 128(M) x 64(N); K step 32.
// BOTH tiles stream through the Tensor Data Mover, double-buffered in LDS,
// one K-step ahead of compute; pad feature produces [rows][40] layout.
// Epilogue modes: 0=f32 store, 1=bf16 TRANSPOSED store (V -> Vt[c][m]),
//                 2=RoPE*0.125 bf16 (Q), 3=RoPE bf16 (K)
// ---------------------------------------------------------------------------
__global__ __launch_bounds__(256)
void gemm_bf16_kernel(const __bf16* __restrict__ A, const __bf16* __restrict__ Wt,
                      int M, int N, int K, int mode,
                      float* __restrict__ outF, __bf16* __restrict__ outH,
                      const float* __restrict__ cosT, const float* __restrict__ sinT) {
#if HAVE_TDM
    __shared__ __align__(16) __bf16 ldsA[2][128][40];  // A tile, double-buffered
    __shared__ __align__(16) __bf16 ldsB[2][64][40];   // B tile, double-buffered
#else
    __shared__ __align__(16) __bf16 ldsB1[64][40];
#endif

    const int tid  = threadIdx.x;
    const int lane = tid & 31;
    const int wave = tid >> 5;
    const int n    = lane & 15;
    const int hi   = lane >> 4;
    const int col0 = blockIdx.x * 64;
    const int row0 = blockIdx.y * 128 + wave * 16;

    v8f acc[4];
#pragma unroll
    for (int t = 0; t < 4; ++t)
#pragma unroll
        for (int r = 0; r < 8; ++r) acc[t][r] = 0.0f;

#if HAVE_TDM
    const __bf16* Ablock = A + (size_t)blockIdx.y * 128 * K;
    const __bf16* Bblock = Wt + (size_t)col0 * K;
    if (wave == 0) {
        tdm_load_2d(Ablock, (unsigned int)(size_t)&ldsA[0][0][0],
                    32u, 128u, (unsigned int)K, (unsigned int)M, (unsigned int)K);
        tdm_load_2d(Bblock, (unsigned int)(size_t)&ldsB[0][0][0],
                    32u, 64u, (unsigned int)K, (unsigned int)N, (unsigned int)K);
    }
#else
    const __bf16* Arow = A + (size_t)(row0 + n) * K;
#endif

    for (int k0 = 0; k0 < K; k0 += 32) {
        const int cur = (k0 >> 5) & 1;
        __syncthreads();   // previous iteration done reading LDS tiles

#if HAVE_TDM
        if (wave == 0) {
            if ((k0 + 32) < K) {
                tdm_load_2d(Ablock + (k0 + 32),
                            (unsigned int)(size_t)&ldsA[cur ^ 1][0][0],
                            32u, 128u, (unsigned int)K, (unsigned int)M,
                            (unsigned int)K);
                tdm_load_2d(Bblock + (k0 + 32),
                            (unsigned int)(size_t)&ldsB[cur ^ 1][0][0],
                            32u, 64u, (unsigned int)K, (unsigned int)N,
                            (unsigned int)K);
                __builtin_amdgcn_s_wait_tensorcnt(2);  // current pair complete
            } else {
                __builtin_amdgcn_s_wait_tensorcnt(0);
            }
        }
        __syncthreads();   // DMA'd tiles visible to all waves

        v16bf a = lda16(&ldsA[cur][wave * 16 + n][0], hi);
#pragma unroll
        for (int t = 0; t < 4; ++t) {
            v16bf b = ldb16(&ldsB[cur][t * 16 + n][16 * hi]);
            acc[t] = wmma_bf16(a, b, acc[t]);
        }
#else
        {   // cooperative B tile fill from Wt (contiguous b128 both sides)
            int rowb  = tid >> 2;          // 0..63
            int chunk = (tid & 3) * 8;     // 0..24
            uint4 w = *(const uint4*)(Wt + (size_t)(col0 + rowb) * K + k0 + chunk);
            if ((k0 + 32) < K)
                __builtin_prefetch(Wt + (size_t)(col0 + rowb) * K + k0 + 32 + chunk, 0, 3);
            *(uint4*)&ldsB1[rowb][chunk] = w;
        }
        __syncthreads();

        v16bf a = lda16(Arow + k0, hi);
#pragma unroll
        for (int t = 0; t < 4; ++t) {
            v16bf b = ldb16(&ldsB1[t * 16 + n][16 * hi]);
            acc[t] = wmma_bf16(a, b, acc[t]);
        }
#endif
    }

    // epilogue
#pragma unroll
    for (int t = 0; t < 4; ++t) {
        int c = col0 + t * 16 + n;
        if (mode == 1) {
            // V path: store transposed, Vt[c][m], 8 rows packed per lane
            Pack8 pk;
#pragma unroll
            for (int r = 0; r < 8; ++r) pk.h[r] = (__bf16)acc[t][r];
            *(uint4*)(outH + (size_t)c * (size_t)M + row0 + 8 * hi) = pk.q;
        } else {
#pragma unroll
            for (int r = 0; r < 8; ++r) {
                int m = row0 + r + 8 * hi;
                float v = acc[t][r];
                if (mode == 0) {
                    outF[(size_t)m * N + c] = v;
                } else {
                    int tpos = m & (TSEQ - 1);
                    int d    = c & (HD - 1);
                    float cs = cosT[tpos * (HD / 2) + (d >> 1)];
                    float sn = sinT[tpos * (HD / 2) + (d >> 1)];
                    float other = __shfl_xor(v, 1, 32);   // partner column c^1
                    float ro = (d & 1) ? (other * sn + v * cs)
                                       : (v * cs - other * sn);
                    if (mode == 2) ro *= QK_SCALE;
                    outH[(size_t)m * N + c] = (__bf16)ro;
                }
            }
        }
    }
}

// ---------------------------------------------------------------------------
// Kernel 4: differential flash attention.
// Grid (T/64, NHEADS, BATCH), block 256 = 8 waves.
// wave -> strip = wave>>1 (16 query rows), half = wave&1 (64 of 128 V cols).
// V is pre-transposed (Vt[c][b*T+s]) so P@V B-fragments load contiguously
// from global: no LDS V staging, no block-wide barriers in the KV loop,
// per-wave causal loop bound.
// ---------------------------------------------------------------------------
__global__ __launch_bounds__(256)
void diff_attn_kernel(const __bf16* __restrict__ Qh, const __bf16* __restrict__ Kh,
                      const __bf16* __restrict__ Vt, __bf16* __restrict__ Ah,
                      const float* __restrict__ lamPtr, const float* __restrict__ g) {
    __shared__ __align__(16) __bf16 ldsP[8][2][16][32];   // per-wave P staging
    __shared__ float ldsS[4][2][16];                      // rms partials

    const int tid   = threadIdx.x;
    const int lane  = tid & 31;
    const int wave  = tid >> 5;
    const int n     = lane & 15;
    const int hi    = lane >> 4;
    const int strip = wave >> 1;
    const int half  = wave & 1;
    const int h     = blockIdx.y;
    const int b     = blockIdx.z;
    const int q0    = blockIdx.x * 64 + strip * 16;
    const float lam = lamPtr[0];

    const size_t rowB = (size_t)b * TSEQ;
    const __bf16* Qbase = Qh + rowB * EMBED;
    const __bf16* Kbase = Kh + rowB * 1024 + h * HD;
    // Vt[c][m]: column base for this head's 128 dims, batch offset in m
    const __bf16* VtBase = Vt + ((size_t)(h >> 1) * DH2) * MROWS + rowB;

    // Q fragments: two q-heads (2h, 2h+1) x two 32-wide K chunks
    v16bf qa[2][2];
    {
        const __bf16* qrow = Qbase + (size_t)(q0 + n) * EMBED;
        qa[0][0] = lda16(qrow + (2 * h) * HD, hi);
        qa[0][1] = lda16(qrow + (2 * h) * HD + 32, hi);
        qa[1][0] = lda16(qrow + (2 * h + 1) * HD, hi);
        qa[1][1] = lda16(qrow + (2 * h + 1) * HD + 32, hi);
    }

    v8f   O[2][4];
    float Mx[2][8], L[2][8];
#pragma unroll
    for (int s = 0; s < 2; ++s)
#pragma unroll
        for (int r = 0; r < 8; ++r) { Mx[s][r] = -3.0e38f; L[s][r] = 0.0f; }
#pragma unroll
    for (int s = 0; s < 2; ++s)
#pragma unroll
        for (int t = 0; t < 4; ++t)
#pragma unroll
            for (int r = 0; r < 8; ++r) O[s][t][r] = 0.0f;

    const int nIterW = (q0 + 47) >> 5;   // per-wave causal bound (keys <= q0+15)

    for (int it = 0; it < nIterW; ++it) {
        const int s0 = it * 32;

        // K^T fragments (contiguous per lane), shared by both score sets
        const __bf16* krow0 = Kbase + (size_t)(s0 + n) * 1024 + 16 * hi;
        const __bf16* krow1 = Kbase + (size_t)(s0 + 16 + n) * 1024 + 16 * hi;
        v16bf bk0a = ldb16(krow0);
        v16bf bk0b = ldb16(krow0 + 32);
        v16bf bk1a = ldb16(krow1);
        v16bf bk1b = ldb16(krow1 + 32);
        if (it + 1 < nIterW)
            __builtin_prefetch(krow0 + 32 * 1024, 0, 3);   // next K tile

#pragma unroll
        for (int set = 0; set < 2; ++set) {
            v8f S0, S1;
#pragma unroll
            for (int r = 0; r < 8; ++r) { S0[r] = 0.0f; S1[r] = 0.0f; }
            S0 = wmma_bf16(qa[set][0], bk0a, S0);
            S0 = wmma_bf16(qa[set][1], bk0b, S0);
            S1 = wmma_bf16(qa[set][0], bk1a, S1);
            S1 = wmma_bf16(qa[set][1], bk1b, S1);

#pragma unroll
            for (int r = 0; r < 8; ++r) {
                int qi = q0 + r + 8 * hi;
                float a0 = (s0 + n      <= qi) ? S0[r] : -3.0e38f;
                float a1 = (s0 + 16 + n <= qi) ? S1[r] : -3.0e38f;
                float rm = fmaxf(a0, a1);
                rm = fmaxf(rm, __shfl_xor(rm, 1, 32));
                rm = fmaxf(rm, __shfl_xor(rm, 2, 32));
                rm = fmaxf(rm, __shfl_xor(rm, 4, 32));
                rm = fmaxf(rm, __shfl_xor(rm, 8, 32));
                float mn = fmaxf(fmaxf(Mx[set][r], rm), -1.0e30f);
                float al = __expf(Mx[set][r] - mn);
                float p0 = __expf(a0 - mn);
                float p1 = __expf(a1 - mn);
                float rs = p0 + p1;
                rs += __shfl_xor(rs, 1, 32);
                rs += __shfl_xor(rs, 2, 32);
                rs += __shfl_xor(rs, 4, 32);
                rs += __shfl_xor(rs, 8, 32);
                L[set][r]  = L[set][r] * al + rs;
                Mx[set][r] = mn;
#pragma unroll
                for (int t = 0; t < 4; ++t) O[set][t][r] *= al;
                ldsP[wave][set][r + 8 * hi][n]      = (__bf16)p0;
                ldsP[wave][set][r + 8 * hi][16 + n] = (__bf16)p1;
            }
        }

        // P @ V: A from per-wave LDS staging, B straight from global Vt
        v16bf pa0 = lda16(&ldsP[wave][0][n][0], hi);
        v16bf pa1 = lda16(&ldsP[wave][1][n][0], hi);
#pragma unroll
        for (int t = 0; t < 4; ++t) {
            const __bf16* vp =
                VtBase + (size_t)(half * 64 + t * 16 + n) * MROWS + s0 + 16 * hi;
            v16bf vb = ldb16(vp);
            if (it + 1 < nIterW)
                __builtin_prefetch(vp + 32, 0, 3);         // next V key block
            O[0][t] = wmma_bf16(pa0, vb, O[0][t]);
            O[1][t] = wmma_bf16(pa1, vb, O[1][t]);
        }
    }

    // epilogue: diff, RMS-norm over 128 dims (cross-wave), gain, store
    float inv0[8], inv1[8];
#pragma unroll
    for (int r = 0; r < 8; ++r) {
        inv0[r] = 1.0f / L[0][r];
        inv1[r] = 1.0f / L[1][r];
    }
    float o[4][8];
    float ssq[8];
#pragma unroll
    for (int r = 0; r < 8; ++r) ssq[r] = 0.0f;
#pragma unroll
    for (int t = 0; t < 4; ++t)
#pragma unroll
        for (int r = 0; r < 8; ++r) {
            float v = O[0][t][r] * inv0[r] - lam * O[1][t][r] * inv1[r];
            o[t][r] = v;
            ssq[r] += v * v;
        }
#pragma unroll
    for (int r = 0; r < 8; ++r) {
        float s = ssq[r];
        s += __shfl_xor(s, 1, 32);
        s += __shfl_xor(s, 2, 32);
        s += __shfl_xor(s, 4, 32);
        s += __shfl_xor(s, 8, 32);
        ssq[r] = s;
    }
    if (n == 0) {
#pragma unroll
        for (int r = 0; r < 8; ++r) ldsS[strip][half][r + 8 * hi] = ssq[r];
    }
    __syncthreads();

#pragma unroll
    for (int t = 0; t < 4; ++t) {
        int c = half * 64 + t * 16 + n;
        float gv = g[c];
#pragma unroll
        for (int r = 0; r < 8; ++r) {
            int row = r + 8 * hi;
            float tot = ldsS[strip][0][row] + ldsS[strip][1][row];
            float rms = sqrtf(tot * (1.0f / 128.0f) + 1e-5f);
            float val = o[t][r] / rms * gv * OUT_SCALE;
            Ah[(rowB + q0 + row) * EMBED + h * DH2 + c] = (__bf16)val;
        }
    }
}

// ---------------------------------------------------------------------------
// Host launcher
// ---------------------------------------------------------------------------
extern "C" void kernel_launch(void* const* d_in, const int* in_sizes, int n_in,
                              void* d_out, int out_size, void* d_ws, size_t ws_size,
                              hipStream_t stream) {
    (void)in_sizes; (void)n_in; (void)out_size; (void)ws_size;

    const float* x    = (const float*)d_in[0];
    const float* cosT = (const float*)d_in[1];
    const float* sinT = (const float*)d_in[2];
    const float* Wq   = (const float*)d_in[3];
    const float* Wk   = (const float*)d_in[4];
    const float* Wv   = (const float*)d_in[5];
    const float* Wo   = (const float*)d_in[6];
    const float* lq1  = (const float*)d_in[7];
    const float* lk1  = (const float*)d_in[8];
    const float* lq2  = (const float*)d_in[9];
    const float* lk2  = (const float*)d_in[10];
    const float* g    = (const float*)d_in[11];
    float* out = (float*)d_out;

    char* ws = (char*)d_ws;
    size_t off = 0;
    auto take = [&](size_t bytes) -> char* {
        char* p = ws + off;
        off = (off + bytes + 255) & ~(size_t)255;
        return p;
    };

    __bf16* Xh   = (__bf16*)take((size_t)MROWS * EMBED * 2);
    __bf16* Wqt  = (__bf16*)take((size_t)EMBED * 2048 * 2);  // 2048 x 2048 (N x K)
    __bf16* Wkt  = (__bf16*)take((size_t)EMBED * 1024 * 2);  // 1024 x 2048
    __bf16* Wvt  = (__bf16*)take((size_t)EMBED * 1024 * 2);  // 1024 x 2048
    __bf16* Wot  = (__bf16*)take((size_t)EMBED * 2048 * 2);  // 2048 x 2048
    __bf16* Qhb  = (__bf16*)take((size_t)MROWS * 2048 * 2);
    __bf16* Khb  = (__bf16*)take((size_t)MROWS * 1024 * 2);
    __bf16* Vtb  = (__bf16*)take((size_t)1024 * MROWS * 2);  // Vt[c][m]
    __bf16* Ahb  = (__bf16*)take((size_t)MROWS * EMBED * 2);
    float*  lamW = (float*)take(256);

    // 1) activations: plain f32->bf16; weights: convert + transpose to N x K
    {
        int nq = (MROWS * EMBED) / 4;
        cvt_bf16_kernel<<<(nq + 255) / 256, 256, 0, stream>>>(x, Xh, nq);
    }
    cvt_transpose_kernel<<<dim3(2048 / 32, EMBED / 32), 256, 0, stream>>>(Wq, Wqt, EMBED, 2048);
    cvt_transpose_kernel<<<dim3(1024 / 32, EMBED / 32), 256, 0, stream>>>(Wk, Wkt, EMBED, 1024);
    cvt_transpose_kernel<<<dim3(1024 / 32, EMBED / 32), 256, 0, stream>>>(Wv, Wvt, EMBED, 1024);
    cvt_transpose_kernel<<<dim3(2048 / 32, EMBED / 32), 256, 0, stream>>>(Wo, Wot, EMBED, 2048);

    // 2) lambda
    lam_kernel<<<1, 64, 0, stream>>>(lq1, lk1, lq2, lk2, lamW);

    // 3) QKV projections with fused RoPE / scaling / transpose epilogues
    gemm_bf16_kernel<<<dim3(2048 / 64, MROWS / 128), 256, 0, stream>>>(
        Xh, Wqt, MROWS, 2048, EMBED, 2, nullptr, Qhb, cosT, sinT);
    gemm_bf16_kernel<<<dim3(1024 / 64, MROWS / 128), 256, 0, stream>>>(
        Xh, Wkt, MROWS, 1024, EMBED, 3, nullptr, Khb, cosT, sinT);
    gemm_bf16_kernel<<<dim3(1024 / 64, MROWS / 128), 256, 0, stream>>>(
        Xh, Wvt, MROWS, 1024, EMBED, 1, nullptr, Vtb, nullptr, nullptr);

    // 4) differential flash attention (fused diff + RMS norm + gain)
    diff_attn_kernel<<<dim3(TSEQ / 64, NHEADS, BATCH), 256, 0, stream>>>(
        Qhb, Khb, Vtb, Ahb, lamW, g);

    // 5) output projection, f32 result
    gemm_bf16_kernel<<<dim3(2048 / 64, MROWS / 128), 256, 0, stream>>>(
        Ahb, Wot, MROWS, 2048, EMBED, 0, out, nullptr, nullptr, nullptr);
}
